// C_SSSWA_87746181857951
// MI455X (gfx1250) — compile-verified, matrix operand order checked by reading, not run on hardware
//
#include <hip/hip_runtime.h>

typedef _Float16 h16;
typedef __attribute__((ext_vector_type(4)))  _Float16 v4h;
typedef __attribute__((ext_vector_type(8)))  _Float16 v8h;
typedef __attribute__((ext_vector_type(16))) _Float16 v16h;
typedef __attribute__((ext_vector_type(4)))  float    v4f;
typedef __attribute__((ext_vector_type(8)))  float    v8f;

#define NWIN 4096
#define EPSF 1e-12f
#define MID_ELEMS (4 * 64 * 256 * 256)   // branch-1 intermediate, f16
// f16 weight arena layout (element offsets past mid)
#define W_QK    0
#define W_V     8192
#define W_PROJ  12288     // [tap][oc][ic] 9*64*64
#define W_QKV   49152
#define W_PROJ1 61440
#define W_TOTAL 98304

__device__ __forceinline__ v8f wmma_f16(v16h a, v16h b, v8f c) {
  return __builtin_amdgcn_wmma_f32_16x16x32_f16(false, a, false, b, (short)0, c,
                                                false, false);
}
__device__ __forceinline__ v16h join16(v8h lo, v8h hi) {
  return __builtin_shufflevector(lo, hi, 0, 1, 2, 3, 4, 5, 6, 7,
                                 8, 9, 10, 11, 12, 13, 14, 15);
}

// A fragment (16x32 MxK) from row-major [m][k] f16 (LDS or global): 2x b128/lane
__device__ __forceinline__ v16h fragA(const h16* S, int ld, int row0, int col0) {
  const int lane = threadIdx.x & 31;
  const h16* p = S + (row0 + (lane & 15)) * ld + col0 + ((lane >> 4) << 3);
  return join16(*(const v8h*)p, *(const v8h*)(p + 16));
}
// B fragment (32x16 KxN) from [n][k] row-major f16 source: 2x b128/lane
__device__ __forceinline__ v16h fragB_T(const h16* S, int ld, int k0, int n0) {
  const int lane = threadIdx.x & 31;
  const h16* p = S + (n0 + (lane & 15)) * ld + k0 + ((lane >> 4) << 3);
  return join16(*(const v8h*)p, *(const v8h*)(p + 16));
}
// B fragment with 3x3 tap shift; S is [pixel][channel] ld=64, zero pad at window edge
__device__ __forceinline__ v16h fragB_shiftT(const h16* S, int k0, int n0,
                                             int dy, int dx) {
  const int lane = threadIdx.x & 31;
  const int pix = n0 + (lane & 15);
  const int py = (pix >> 3) + dy, px = (pix & 7) + dx;
  const bool ok = (py >= 0) && (py < 8) && (px >= 0) && (px < 8);
  const int sp = ok ? (py * 8 + px) : 0;
  const h16* p = S + sp * 64 + k0 + ((lane >> 4) << 3);
  const v8h z = {};
  const v8h lo = ok ? *(const v8h*)p : z;
  const v8h hi = ok ? *(const v8h*)(p + 16) : z;
  return join16(lo, hi);
}
// transposed D store: each lane writes one contiguous 8-half run (b128)
__device__ __forceinline__ void storeD_hT(h16* D, int ldT, int row0, int col0, v8f acc) {
  const int lane = threadIdx.x & 31;
  const int n = col0 + (lane & 15);
  const int mb = row0 + ((lane >> 4) << 3);
  v8h o;
#pragma unroll
  for (int r = 0; r < 8; ++r) o[r] = (h16)acc[r];
  *(v8h*)(D + n * ldT + mb) = o;
}
__device__ __forceinline__ void storeD_fT(float* D, int ldT, int row0, int col0,
                                          v8f acc, float s) {
  const int lane = threadIdx.x & 31;
  const int n = col0 + (lane & 15);
  const int mb = row0 + ((lane >> 4) << 3);
  v4f a, b;
#pragma unroll
  for (int r = 0; r < 4; ++r) { a[r] = acc[r] * s; b[r] = acc[r + 4] * s; }
  *(v4f*)(D + n * ldT + mb) = a;
  *(v4f*)(D + n * ldT + mb + 4) = b;
}

// ---------------- weight prep: f32 -> f16, proj re-laid [tap][oc][ic] --------
__global__ __launch_bounds__(256) void prep_weights(
    const float* __restrict__ qk_w, const float* __restrict__ v_w,
    const float* __restrict__ proj_w, const float* __restrict__ qkv_w,
    const float* __restrict__ proj1_w, h16* __restrict__ wb) {
  const int i = blockIdx.x * 256 + threadIdx.x;
  if (i >= W_TOTAL) return;
  float v;
  if (i < 8192) v = qk_w[i];
  else if (i < 12288) v = v_w[i - 8192];
  else if (i < 49152) {
    const int r = i - 12288, tap = r >> 12, mk = r & 4095;
    v = proj_w[(mk >> 6) * 576 + (mk & 63) * 9 + tap];
  } else if (i < 61440) v = qkv_w[i - 49152];
  else {
    const int r = i - 61440, tap = r >> 12, mk = r & 4095;
    v = proj1_w[(mk >> 6) * 576 + (mk & 63) * 9 + tap];
  }
  wb[i] = (h16)v;
}

// ======================= Kernel 1: branch 1 (un-shifted) =====================
// window n <-> (b=n>>10, ch=(n>>4)&63, ihp=(n&15)*2); slot cc <-> (ih parity, iw)
__global__ __launch_bounds__(256) void swa_branch1(
    const float* __restrict__ x, const float* __restrict__ xf,
    const float* __restrict__ temperature,
    const float* __restrict__ qk_dw, const float* __restrict__ v_dw,
    const h16* __restrict__ wb, h16* __restrict__ mid) {
  extern __shared__ char smem_raw[];
  h16* rA = (h16*)smem_raw;     // 4096: xf [p][c] -> q [c][p] -> out [p][c]
  h16* rB = rA + 4096;          // 4096: x  [p][c] -> k [c][p]
  h16* rC = rB + 4096;          // 8192: qk1x1 [p][2c] -> logits f32 [kc][qc]
  h16* rD = rC + 8192;          // 4096: v1x1 [p][c] -> attn probs [qc][kc]
  h16* rE = rD + 4096;          // 4096: v [p][c]
  float* wdw = (float*)(rE + 4096);  // 1152 qk-dw [tap][c] + 576 v-dw [tap][c]
  float* attnT = (float*)rC;
  float* wdwV = wdw + 1152;

  const int n = blockIdx.x;
  const int tid = threadIdx.x;
  const int w = tid >> 5;
  const int b = n >> 10;
  const int ch = (n >> 4) & 63;
  const int ihp = (n & 15) << 1;
  const float invT = 1.0f / temperature[0];

  // ---- stage 0: vectorized gather (torch-view scrambled map) + dw weights ----
  for (int t = tid; t < 1024; t += 256) {
    const int cc = t >> 4, q = t & 15;
    const int py = q >> 1, px0 = (q & 1) << 2;
    const int gy = (ihp | (cc >> 5)) * 8 + py;
    const int gx = (cc & 31) * 8 + px0;
    const int g = ((b * 64 + ch) * 256 + gy) * 256 + gx;
    const v4f vf = *(const v4f*)(xf + g);
    const v4f vx = *(const v4f*)(x + g);
#pragma unroll
    for (int u = 0; u < 4; ++u) {
      rA[(py * 8 + px0 + u) * 64 + cc] = (h16)vf[u];
      rB[(py * 8 + px0 + u) * 64 + cc] = (h16)vx[u];
    }
  }
  for (int i = tid; i < 1152; i += 256) wdw[(i % 9) * 128 + (i / 9)] = qk_dw[i];
  for (int i = tid; i < 576; i += 256) wdwV[(i % 9) * 64 + (i / 9)] = v_dw[i];
  __syncthreads();

  // ---- stage 1: 1x1 convs as GEMMs (A = f16 weights in global arena) ----
#pragma unroll
  for (int t4 = 0; t4 < 4; ++t4) {                 // qk: 128x64 = 32 tiles
    const int t = w * 4 + t4, tm = t >> 2, tn = t & 3;
    v8f acc = {};
#pragma unroll
    for (int ks = 0; ks < 64; ks += 32)
      acc = wmma_f16(fragA(wb + W_QK, 64, tm * 16, ks),
                     fragB_T(rA, 64, ks, tn * 16), acc);
    storeD_hT(rC, 128, tm * 16, tn * 16, acc);
  }
#pragma unroll
  for (int t2 = 0; t2 < 2; ++t2) {                 // v: 64x64 = 16 tiles
    const int t = w * 2 + t2, tm = t >> 2, tn = t & 3;
    v8f acc = {};
#pragma unroll
    for (int ks = 0; ks < 64; ks += 32)
      acc = wmma_f16(fragA(wb + W_V, 64, tm * 16, ks),
                     fragB_T(rB, 64, ks, tn * 16), acc);
    storeD_hT(rD, 64, tm * 16, tn * 16, acc);
  }
  __syncthreads();

  // ---- stage 2: depthwise 3x3, channel-octet per pixel (b128 activation loads)
  for (int t = tid; t < 1024; t += 256) {          // qk: 128ch x 64px / 8
    const int pp = t >> 4, c8 = (t & 15) << 3;
    const int py = pp >> 3, px = pp & 7;
    float a[8];
#pragma unroll
    for (int u = 0; u < 8; ++u) a[u] = 0.f;
#pragma unroll
    for (int dy = -1; dy <= 1; ++dy)
#pragma unroll
      for (int dx = -1; dx <= 1; ++dx) {
        const int sy = py + dy, sx = px + dx;
        if (sy >= 0 && sy < 8 && sx >= 0 && sx < 8) {
          const int tap = (dy + 1) * 3 + (dx + 1);
          const v8h xv = *(const v8h*)(rC + (sy * 8 + sx) * 128 + c8);
          const float* wp = wdw + tap * 128 + c8;
#pragma unroll
          for (int u = 0; u < 8; ++u) a[u] += wp[u] * (float)xv[u];
        }
      }
    h16* dst = (c8 < 64) ? rA : rB;
    const int cb = (c8 < 64) ? c8 : (c8 - 64);
#pragma unroll
    for (int u = 0; u < 8; ++u) dst[(cb + u) * 64 + pp] = (h16)a[u];
  }
  for (int t = tid; t < 512; t += 256) {           // v: 64ch x 64px / 8
    const int pp = t >> 3, c8 = (t & 7) << 3;
    const int py = pp >> 3, px = pp & 7;
    float a[8];
#pragma unroll
    for (int u = 0; u < 8; ++u) a[u] = 0.f;
#pragma unroll
    for (int dy = -1; dy <= 1; ++dy)
#pragma unroll
      for (int dx = -1; dx <= 1; ++dx) {
        const int sy = py + dy, sx = px + dx;
        if (sy >= 0 && sy < 8 && sx >= 0 && sx < 8) {
          const int tap = (dy + 1) * 3 + (dx + 1);
          const v8h xv = *(const v8h*)(rD + (sy * 8 + sx) * 64 + c8);
          const float* wp = wdwV + tap * 64 + c8;
#pragma unroll
          for (int u = 0; u < 8; ++u) a[u] += wp[u] * (float)xv[u];
        }
      }
    v8h o;
#pragma unroll
    for (int u = 0; u < 8; ++u) o[u] = (h16)a[u];
    *(v8h*)(rE + pp * 64 + c8) = o;
  }
  __syncthreads();

  // ---- stage 3: L2-normalize q,k rows (vectorized row passes) ----
  if (tid < 128) {
    h16* R = (tid < 64) ? (rA + tid * 64) : (rB + (tid - 64) * 64);
    float s = 0.f;
#pragma unroll
    for (int j = 0; j < 8; ++j) {
      const v8h v = *(const v8h*)(R + j * 8);
#pragma unroll
      for (int u = 0; u < 8; ++u) { const float f = (float)v[u]; s += f * f; }
    }
    const float inv = 1.0f / fmaxf(sqrtf(s), EPSF);
#pragma unroll
    for (int j = 0; j < 8; ++j) {
      const v8h v = *(const v8h*)(R + j * 8);
      v8h o;
#pragma unroll
      for (int u = 0; u < 8; ++u) o[u] = (h16)((float)v[u] * inv);
      *(v8h*)(R + j * 8) = o;
    }
  }
  __syncthreads();

  // ---- stage 4: logits = Qn @ Kn^T, stored transposed [kc][qc] f32 ----
#pragma unroll
  for (int t2 = 0; t2 < 2; ++t2) {
    const int t = w * 2 + t2, tm = t >> 2, tn = t & 3;
    v8f acc = {};
#pragma unroll
    for (int ks = 0; ks < 64; ks += 32)
      acc = wmma_f16(fragA(rA, 64, tm * 16, ks),
                     fragB_T(rB, 64, ks, tn * 16), acc);
    storeD_fT(attnT, 64, tm * 16, tn * 16, acc, invT);
  }
  __syncthreads();

  // ---- stage 5: softmax (conflict-free column reads) -> probs [qc][kc] f16 ----
  if (tid < 64) {
    float mx = -3.4e38f;
    for (int j = 0; j < 64; ++j) mx = fmaxf(mx, attnT[j * 64 + tid]);
    float s = 0.f;
    for (int j = 0; j < 64; ++j) {
      const float e = __expf(attnT[j * 64 + tid] - mx);
      attnT[j * 64 + tid] = e; s += e;
    }
    const float inv = 1.0f / s;
#pragma unroll
    for (int j8 = 0; j8 < 8; ++j8) {
      v8h o;
#pragma unroll
      for (int u = 0; u < 8; ++u) o[u] = (h16)(attnT[(j8 * 8 + u) * 64 + tid] * inv);
      *(v8h*)(rD + tid * 64 + j8 * 8) = o;
    }
  }
  __syncthreads();

  // ---- stage 6: out = attn @ V  ([p][c] result for the conv stage) ----
#pragma unroll
  for (int t2 = 0; t2 < 2; ++t2) {
    const int t = w * 2 + t2, tm = t >> 2, tn = t & 3;
    v8f acc = {};
#pragma unroll
    for (int ks = 0; ks < 64; ks += 32)
      acc = wmma_f16(fragA(rD, 64, tm * 16, ks),
                     fragB_T(rE, 64, ks, tn * 16), acc);
    storeD_hT(rA, 64, tm * 16, tn * 16, acc);
  }
  __syncthreads();

  // ---- stage 7: dense 3x3 proj = 9 shifted GEMMs; scatter (window reverse) ----
#pragma unroll
  for (int t2 = 0; t2 < 2; ++t2) {
    const int t = w * 2 + t2, tm = t >> 2, tn = t & 3;
    v8f acc = {};
#pragma unroll
    for (int tap = 0; tap < 9; ++tap) {
      const int dy = tap / 3 - 1, dx = tap % 3 - 1;
#pragma unroll
      for (int ks = 0; ks < 64; ks += 32)
        acc = wmma_f16(fragA(wb + W_PROJ + tap * 4096, 64, tm * 16, ks),
                       fragB_shiftT(rA, ks, tn * 16, dy, dx), acc);
    }
    const int lane = tid & 31;
    const int p = tn * 16 + (lane & 15);
    const int mb = tm * 16 + ((lane >> 4) << 3);
    const int py = p >> 3, px = p & 7;
#pragma unroll
    for (int r = 0; r < 8; ++r) {
      const int cc = mb + r;
      const int gy = (ihp | (cc >> 5)) * 8 + py;
      const int gx = (cc & 31) * 8 + px;
      mid[((b * 64 + ch) * 256 + gy) * 256 + gx] = (h16)acc[r];
    }
  }
}

// ================= Kernel 2: branch 2 (shifted windows, fused qkv) ===========
__global__ __launch_bounds__(256) void swa_branch2(
    const h16* __restrict__ mid, const float* __restrict__ temperature,
    const float* __restrict__ qkv_dw, const h16* __restrict__ wb,
    float* __restrict__ out) {
  extern __shared__ char smem_raw[];
  h16* rA = (h16*)smem_raw;     // 4096: in [p][c] -> q [c][p] -> out [p][c]
  h16* rB = rA + 4096;          // 12288: qkv1x1 [p][3c] -> logits f32 + probs
  h16* rC = rB + 12288;         // 4096: k [c][p]
  h16* rD = rC + 4096;          // 4096: v [p][c]
  float* wdw = (float*)(rD + 4096);  // 1728: [tap][c]
  float* attnT = (float*)rB;
  h16* bufAh = rB + 8192;

  const int n = blockIdx.x;
  const int tid = threadIdx.x;
  const int w = tid >> 5;
  const int b = n >> 10;
  const int ch = (n >> 4) & 63;
  const int ihp = (n & 15) << 1;
  const float invT = 1.0f / temperature[0];

  // ---- gather from mid, roll(-4,-4) folded into index math ----
  for (int t = tid; t < 1024; t += 256) {
    const int cc = t >> 4, q = t & 15;
    const int py = q >> 1, px0 = (q & 1) << 2;
    const int gy = ((ihp | (cc >> 5)) * 8 + py + 4) & 255;
    const int gx = ((cc & 31) * 8 + px0 + 4) & 255;
    const v4h m4 = *(const v4h*)(mid + ((b * 64 + ch) * 256 + gy) * 256 + gx);
#pragma unroll
    for (int u = 0; u < 4; ++u) rA[(py * 8 + px0 + u) * 64 + cc] = m4[u];
  }
  for (int i = tid; i < 1728; i += 256) wdw[(i % 9) * 192 + (i / 9)] = qkv_dw[i];
  __syncthreads();

  // ---- qkv 1x1: 192x64 = 48 tiles, 6 per wave ----
#pragma unroll
  for (int t6 = 0; t6 < 6; ++t6) {
    const int t = w * 6 + t6, tm = t >> 2, tn = t & 3;
    v8f acc = {};
#pragma unroll
    for (int ks = 0; ks < 64; ks += 32)
      acc = wmma_f16(fragA(wb + W_QKV, 64, tm * 16, ks),
                     fragB_T(rA, 64, ks, tn * 16), acc);
    storeD_hT(rB, 192, tm * 16, tn * 16, acc);
  }
  __syncthreads();

  // ---- depthwise 3x3 over 192 channels ----
  for (int t = tid; t < 1536; t += 256) {
    const int pp = t / 24, c8 = (t % 24) << 3;
    const int py = pp >> 3, px = pp & 7;
    float a[8];
#pragma unroll
    for (int u = 0; u < 8; ++u) a[u] = 0.f;
#pragma unroll
    for (int dy = -1; dy <= 1; ++dy)
#pragma unroll
      for (int dx = -1; dx <= 1; ++dx) {
        const int sy = py + dy, sx = px + dx;
        if (sy >= 0 && sy < 8 && sx >= 0 && sx < 8) {
          const int tap = (dy + 1) * 3 + (dx + 1);
          const v8h xv = *(const v8h*)(rB + (sy * 8 + sx) * 192 + c8);
          const float* wp = wdw + tap * 192 + c8;
#pragma unroll
          for (int u = 0; u < 8; ++u) a[u] += wp[u] * (float)xv[u];
        }
      }
    if (c8 >= 128) {
      v8h o;
#pragma unroll
      for (int u = 0; u < 8; ++u) o[u] = (h16)a[u];
      *(v8h*)(rD + pp * 64 + (c8 - 128)) = o;
    } else {
      h16* dst = (c8 < 64) ? rA : rC;
      const int cb = (c8 < 64) ? c8 : (c8 - 64);
#pragma unroll
      for (int u = 0; u < 8; ++u) dst[(cb + u) * 64 + pp] = (h16)a[u];
    }
  }
  __syncthreads();

  // ---- normalize q,k ----
  if (tid < 128) {
    h16* R = (tid < 64) ? (rA + tid * 64) : (rC + (tid - 64) * 64);
    float s = 0.f;
#pragma unroll
    for (int j = 0; j < 8; ++j) {
      const v8h v = *(const v8h*)(R + j * 8);
#pragma unroll
      for (int u = 0; u < 8; ++u) { const float f = (float)v[u]; s += f * f; }
    }
    const float inv = 1.0f / fmaxf(sqrtf(s), EPSF);
#pragma unroll
    for (int j = 0; j < 8; ++j) {
      const v8h v = *(const v8h*)(R + j * 8);
      v8h o;
#pragma unroll
      for (int u = 0; u < 8; ++u) o[u] = (h16)((float)v[u] * inv);
      *(v8h*)(R + j * 8) = o;
    }
  }
  __syncthreads();

  // ---- logits -> [kc][qc] f32 ----
#pragma unroll
  for (int t2 = 0; t2 < 2; ++t2) {
    const int t = w * 2 + t2, tm = t >> 2, tn = t & 3;
    v8f acc = {};
#pragma unroll
    for (int ks = 0; ks < 64; ks += 32)
      acc = wmma_f16(fragA(rA, 64, tm * 16, ks),
                     fragB_T(rC, 64, ks, tn * 16), acc);
    storeD_fT(attnT, 64, tm * 16, tn * 16, acc, invT);
  }
  __syncthreads();

  // ---- softmax ----
  if (tid < 64) {
    float mx = -3.4e38f;
    for (int j = 0; j < 64; ++j) mx = fmaxf(mx, attnT[j * 64 + tid]);
    float s = 0.f;
    for (int j = 0; j < 64; ++j) {
      const float e = __expf(attnT[j * 64 + tid] - mx);
      attnT[j * 64 + tid] = e; s += e;
    }
    const float inv = 1.0f / s;
#pragma unroll
    for (int j8 = 0; j8 < 8; ++j8) {
      v8h o;
#pragma unroll
      for (int u = 0; u < 8; ++u) o[u] = (h16)(attnT[(j8 * 8 + u) * 64 + tid] * inv);
      *(v8h*)(bufAh + tid * 64 + j8 * 8) = o;
    }
  }
  __syncthreads();

  // ---- attn @ V ----
#pragma unroll
  for (int t2 = 0; t2 < 2; ++t2) {
    const int t = w * 2 + t2, tm = t >> 2, tn = t & 3;
    v8f acc = {};
#pragma unroll
    for (int ks = 0; ks < 64; ks += 32)
      acc = wmma_f16(fragA(bufAh, 64, tm * 16, ks),
                     fragB_T(rD, 64, ks, tn * 16), acc);
    storeD_hT(rA, 64, tm * 16, tn * 16, acc);
  }
  __syncthreads();

  // ---- dense 3x3 proj1; scatter with roll(+4,+4) applied ----
#pragma unroll
  for (int t2 = 0; t2 < 2; ++t2) {
    const int t = w * 2 + t2, tm = t >> 2, tn = t & 3;
    v8f acc = {};
#pragma unroll
    for (int tap = 0; tap < 9; ++tap) {
      const int dy = tap / 3 - 1, dx = tap % 3 - 1;
#pragma unroll
      for (int ks = 0; ks < 64; ks += 32)
        acc = wmma_f16(fragA(wb + W_PROJ1 + tap * 4096, 64, tm * 16, ks),
                       fragB_shiftT(rA, ks, tn * 16, dy, dx), acc);
    }
    const int lane = tid & 31;
    const int p = tn * 16 + (lane & 15);
    const int mb = tm * 16 + ((lane >> 4) << 3);
    const int py = p >> 3, px = p & 7;
#pragma unroll
    for (int r = 0; r < 8; ++r) {
      const int cc = mb + r;
      const int gy = ((ihp | (cc >> 5)) * 8 + py + 4) & 255;
      const int gx = ((cc & 31) * 8 + px + 4) & 255;
      out[((b * 64 + ch) * 256 + gy) * 256 + gx] = acc[r];
    }
  }
}

extern "C" void kernel_launch(void* const* d_in, const int* in_sizes, int n_in,
                              void* d_out, int out_size, void* d_ws, size_t ws_size,
                              hipStream_t stream) {
  const float* x       = (const float*)d_in[0];
  const float* x_f     = (const float*)d_in[1];
  const float* temp    = (const float*)d_in[2];
  const float* qk_w    = (const float*)d_in[3];
  const float* qk_dw   = (const float*)d_in[4];
  const float* v_w     = (const float*)d_in[5];
  const float* v_dw    = (const float*)d_in[6];
  const float* proj_w  = (const float*)d_in[7];
  const float* proj1_w = (const float*)d_in[8];
  const float* qkv1_w  = (const float*)d_in[9];
  const float* qkv1_dw = (const float*)d_in[10];
  (void)in_sizes; (void)n_in; (void)out_size; (void)ws_size;

  h16* mid = (h16*)d_ws;                 // 32 MiB f16 intermediate
  h16* wb  = mid + MID_ELEMS;            // 192 KiB f16 weight arena
  float* o = (float*)d_out;

  const size_t smem = 24576u * sizeof(h16) + 1728u * sizeof(float);  // 56064 B

  prep_weights<<<(W_TOTAL + 255) / 256, 256, 0, stream>>>(qk_w, v_w, proj_w,
                                                          qkv1_w, proj1_w, wb);
  swa_branch1<<<NWIN, 256, smem, stream>>>(x, x_f, temp, qk_dw, v_dw, wb, mid);
  swa_branch2<<<NWIN, 256, smem, stream>>>(mid, temp, qkv1_dw, wb, o);
}
